// VanillaCodebook_39702677684264
// MI455X (gfx1250) — matrix operand order repaired and checked by reading, not compiled, not run on hardware
//
#include <hip/hip_runtime.h>

// ---------------------------------------------------------------------------
// VanillaCodebook nearest-neighbor quantization for MI455X (gfx1250).
//   argmin_k ||x - c_k||^2  ==  argmax_k (x . c_k - 0.5 ||c_k||^2)
// GEMM (262144 x 64 x 1024) on v_wmma_f32_16x16x32_bf16. Codebook staged as
// bf16 into LDS by the Tensor Data Mover (tensor_load_to_lds + TENSORcnt),
// with TDM LDS-padding producing a stride-36-dword layout so the 16 column
// lanes of each ds_load_b128 hit all 64 LDS banks conflict-free.
// ---------------------------------------------------------------------------

typedef __attribute__((ext_vector_type(16))) __bf16   v16bf;
typedef __attribute__((ext_vector_type(8)))  float    v8f;
typedef __attribute__((ext_vector_type(4)))  unsigned v4u;
typedef __attribute__((ext_vector_type(8)))  int      v8i;
typedef __attribute__((ext_vector_type(4)))  int      v4i;

#define NUM_NT   1024
#define DIM      64
#define M_TOTAL  (64 * 4096)               // 262144 rows
#define WG_THREADS 256
#define ROWS_PER_WAVE 16
#define ROWS_PER_WG   128                  // 8 waves * 16 rows
#define NBLOCKS  (M_TOTAL / ROWS_PER_WG)   // 2048 row-blocks
#define CBK_DW      (NUM_NT * DIM / 2)     // 32768 dwords packed bf16 (ws)
#define ROW_DW      32                     // packed dwords per codebook row
#define ROW_DW_PAD  36                     // LDS row stride (bank-conflict-free)
#define CBK_LDS_DW  (NUM_NT * ROW_DW_PAD)  // 36864 dwords = 144 KB
#define LDS_BYTES   (CBK_LDS_DW * 4 + NUM_NT * 4)   // 151552 B (2 WG/WGP)

union U8V {
    unsigned u[8];
    v16bf    v;
};

// f32 pair -> packed bf16 (RNE); lowers to v_cvt_pk_bf16_f32
__device__ inline unsigned pack_bf16x2(float lo, float hi) {
    union { __bf16 h[2]; unsigned u; } x;
    x.h[0] = (__bf16)lo;
    x.h[1] = (__bf16)hi;
    return x.u;
}

// order-preserving map: float bits -> unsigned with same ordering
__device__ inline unsigned mono_f32(float f) {
    unsigned u = __float_as_uint(f);
    return (u & 0x80000000u) ? ~u : (u | 0x80000000u);
}

// ---------------------------------------------------------------------------
// TDM staging: 1D tensor of dwords, optional LDS padding (32 dw + 4 dw pad).
// D# layout per CDNA5 ISA ch.10.8 (group0: count/lds/global/type,
// group1: data_size|pad ctl, dims, tile, strides).
// ---------------------------------------------------------------------------
#if __has_builtin(__builtin_amdgcn_tensor_load_to_lds)
#define HAVE_TDM 1
__device__ inline void tdm_load_1d(unsigned lds_off, const void* gptr,
                                   unsigned ndw, int pad) {
    unsigned long long ga = (unsigned long long)(size_t)gptr;
    v4u g0;
    g0[0] = 1u;                                     // count=1, user descriptor
    g0[1] = lds_off;                                // LDS byte address
    g0[2] = (unsigned)ga;                           // global_addr[31:0]
    g0[3] = ((unsigned)(ga >> 32) & 0x01ffffffu)    // global_addr[56:32]
            | 0x80000000u;                          // type=2 ("image")
    unsigned ctl = (2u << 16);                      // data_size = 4 bytes
    if (pad) {
        // pad_enable | pad_interval=code4 (32 dw) | pad_amount=code3 (4 dw)
        ctl |= (1u << 20) | (4u << 22) | (3u << 25);
    }
    v8i g1;
    g1[0] = (int)ctl;                               // mask=0 | ctl
    g1[1] = (int)(ndw << 16);                       // tensor_dim0[15:0]
    g1[2] = (int)((ndw >> 16) | (1u << 16));        // tensor_dim0 hi | dim1=1
    g1[3] = (int)(ndw << 16);                       // tile_dim0
    g1[4] = 1;                                      // tile_dim1=1, tile_dim2=0
    g1[5] = (int)ndw;                               // tensor_dim0_stride lo
    g1[6] = (int)((ndw & 0xffffu) << 16);           // stride0 hi=0 | stride1 lo
    g1[7] = (int)(ndw >> 16);                       // stride1 hi
    v4i z4 = {0, 0, 0, 0};
#if __has_include(<hip/amd_detail/amd_gfx1250_TDM.h>)
    v8i z8 = {0, 0, 0, 0, 0, 0, 0, 0};
    __builtin_amdgcn_tensor_load_to_lds(g0, g1, z4, z4, z8, 0);
#else
    __builtin_amdgcn_tensor_load_to_lds(g0, g1, z4, z4, 0);
#endif
}
#else
#define HAVE_TDM 0
#endif

// ---------------------------------------------------------------------------
// Prep: codebook fp32 -> packed bf16 rows (linear) + 0.5*||c||^2 into d_ws.
// ---------------------------------------------------------------------------
__global__ void vq_prep_codebook(const float* __restrict__ cb,
                                 unsigned* __restrict__ cbk_bf16,
                                 float* __restrict__ hcsq) {
    int row = blockIdx.x * blockDim.x + threadIdx.x;
    if (row >= NUM_NT) return;
    const float* r = cb + (size_t)row * DIM;
    float s = 0.f;
    unsigned* out = cbk_bf16 + (size_t)row * ROW_DW;
#pragma unroll
    for (int i = 0; i < ROW_DW; ++i) {
        float a = r[2 * i], b = r[2 * i + 1];
        s += a * a + b * b;
        out[i] = pack_bf16x2(a, b);
    }
    hcsq[row] = 0.5f * s;
}

// ---------------------------------------------------------------------------
// Main kernel.
// ---------------------------------------------------------------------------
__global__ __launch_bounds__(WG_THREADS, 2)
void vq_argmin_gather(const float* __restrict__ codes,
                      const float* __restrict__ codebook,
                      const unsigned* __restrict__ cbk_bf16,
                      const float* __restrict__ hcsq_g,
                      float* __restrict__ out_codes,
                      float* __restrict__ out_ids) {
    extern __shared__ unsigned smem[];
    unsigned* cbk  = smem;                          // stride-36 padded rows
    float*    hcsq = (float*)(smem + CBK_LDS_DW);   // 1024 floats

    const int lane = threadIdx.x & 31;
    const int wave = threadIdx.x >> 5;
    const int sub  = lane & 15;   // WMMA column (n%16) / A-row (m%16)
    const int kh   = lane >> 4;   // K-half selector in WMMA operand layouts

    // ---- stage codebook into LDS ----
#if HAVE_TDM
    if (wave == 0) {
        unsigned base = (unsigned)(size_t)(void*)smem;
        tdm_load_1d(base, cbk_bf16, CBK_DW, /*pad=*/1);          // 128->144 KB
        tdm_load_1d(base + CBK_LDS_DW * 4, hcsq_g, NUM_NT, 0);   // 4 KB
#if __has_builtin(__builtin_amdgcn_s_wait_tensorcnt)
        __builtin_amdgcn_s_wait_tensorcnt(0);
#else
        asm volatile("s_wait_tensorcnt 0x0" ::: "memory");
#endif
    }
#else
    {
        const uint4* src = (const uint4*)cbk_bf16;
        for (int i = threadIdx.x; i < CBK_DW / 4; i += WG_THREADS) {
            int row = i >> 3, w = i & 7;            // 8 uint4 per 32-dw row
            *(uint4*)(cbk + row * ROW_DW_PAD + w * 4) = src[i];
        }
        for (int i = threadIdx.x; i < NUM_NT; i += WG_THREADS)
            hcsq[i] = hcsq_g[i];
    }
#endif
    __syncthreads();

    for (int blk = blockIdx.x; blk < NBLOCKS; blk += gridDim.x) {
        const int row_base = blk * ROWS_PER_WG + wave * ROWS_PER_WAVE;

        // ---- A operands (bf16 16x32): lane sub = row m; per lane both
        // 16-byte chunks (VGPR0-3: K=kh*16.., VGPR4-7: K=kh*16+8..) are
        // contiguous fp32 spans of the row.
        const float* arow = codes + (size_t)(row_base + sub) * DIM;
        U8V a0, a1;
        {
            const int k = kh * 16;
#pragma unroll
            for (int bk = 0; bk < 2; ++bk) {
                U8V& A = bk ? a1 : a0;
                const int kb = bk * 32 + k;
                float4 f0 = *(const float4*)(arow + kb);
                float4 f1 = *(const float4*)(arow + kb + 4);
                float4 f2 = *(const float4*)(arow + kb + 8);
                float4 f3 = *(const float4*)(arow + kb + 12);
                A.u[0] = pack_bf16x2(f0.x, f0.y);
                A.u[1] = pack_bf16x2(f0.z, f0.w);
                A.u[2] = pack_bf16x2(f1.x, f1.y);
                A.u[3] = pack_bf16x2(f1.z, f1.w);
                A.u[4] = pack_bf16x2(f2.x, f2.y);
                A.u[5] = pack_bf16x2(f2.z, f2.w);
                A.u[6] = pack_bf16x2(f3.x, f3.y);
                A.u[7] = pack_bf16x2(f3.z, f3.w);
            }
        }

        float bestv[8];
        int   besti[8];
#pragma unroll
        for (int r = 0; r < 8; ++r) { bestv[r] = -3.4e38f; besti[r] = 0; }

        // ---- sweep 64 codebook tiles, two per iteration (independent WMMA
        // chains hide the bf16 WMMA->VALU hazard window) ----
        for (int t = 0; t < 64; t += 2) {
            const int n0 = t * 16 + sub;
            const int n1 = n0 + 16;
            const unsigned* p0 = cbk + n0 * ROW_DW_PAD + kh * 8;
            const unsigned* p1 = cbk + n1 * ROW_DW_PAD + kh * 8;
            U8V b00, b01, b10, b11;
            *(uint4*)&b00.u[0] = *(const uint4*)(p0);
            *(uint4*)&b00.u[4] = *(const uint4*)(p0 + 4);
            *(uint4*)&b01.u[0] = *(const uint4*)(p0 + 16);
            *(uint4*)&b01.u[4] = *(const uint4*)(p0 + 20);
            *(uint4*)&b10.u[0] = *(const uint4*)(p1);
            *(uint4*)&b10.u[4] = *(const uint4*)(p1 + 4);
            *(uint4*)&b11.u[0] = *(const uint4*)(p1 + 16);
            *(uint4*)&b11.u[4] = *(const uint4*)(p1 + 20);

            v8f acc0 = {}, acc1 = {};
            acc0 = __builtin_amdgcn_wmma_f32_16x16x32_bf16(
                       false, a0.v, false, b00.v, (short)0, acc0, false, false);
            acc1 = __builtin_amdgcn_wmma_f32_16x16x32_bf16(
                       false, a0.v, false, b10.v, (short)0, acc1, false, false);
            acc0 = __builtin_amdgcn_wmma_f32_16x16x32_bf16(
                       false, a1.v, false, b01.v, (short)0, acc0, false, false);
            acc1 = __builtin_amdgcn_wmma_f32_16x16x32_bf16(
                       false, a1.v, false, b11.v, (short)0, acc1, false, false);

            const float h0 = hcsq[n0];
            const float h1 = hcsq[n1];
#pragma unroll
            for (int r = 0; r < 8; ++r) {
                float s0 = acc0[r] - h0;
                if (s0 > bestv[r]) { bestv[r] = s0; besti[r] = n0; }
                float s1 = acc1[r] - h1;
                if (s1 > bestv[r]) { bestv[r] = s1; besti[r] = n1; }
            }
        }

        // ---- branchless argmax butterfly over the 16-lane column group:
        // 64-bit key = mono(val) << 10 | (1023 - idx)  (max => min idx ties)
        int fi[8];
#pragma unroll
        for (int r = 0; r < 8; ++r) {
            unsigned long long key =
                ((unsigned long long)mono_f32(bestv[r]) << 10)
                | (unsigned)(1023 - besti[r]);
#pragma unroll
            for (int off = 1; off < 16; off <<= 1) {
                unsigned long long ok = __shfl_xor(key, off);
                key = (ok > key) ? ok : key;
            }
            fi[r] = 1023 - (int)(key & 1023u);
        }

        // ---- quant_id (exact small ints as fp32) ----
        {
            int myid = 0;
#pragma unroll
            for (int r = 0; r < 8; ++r) myid = (sub == r) ? fi[r] : myid;
            if (sub < 8) out_ids[row_base + kh * 8 + sub] = (float)myid;
        }

        // ---- quant_codes gather: 256 B/row, 8 B/lane, coalesced stores ----
#pragma unroll
        for (int m = 0; m < 16; ++m) {
            int idx = __shfl(fi[m & 7], (m >> 3) * 16);
            const float2* srcp = (const float2*)(codebook + (size_t)idx * DIM);
            float2*       dstp = (float2*)(out_codes + (size_t)(row_base + m) * DIM);
            dstp[lane] = srcp[lane];
        }
    }
}

// ---------------------------------------------------------------------------
extern "C" void kernel_launch(void* const* d_in, const int* in_sizes, int n_in,
                              void* d_out, int out_size, void* d_ws, size_t ws_size,
                              hipStream_t stream) {
    const float* codes    = (const float*)d_in[0];   // (64, 4096, 64) fp32
    const float* codebook = (const float*)d_in[1];   // (1024, 64) fp32

    unsigned* cbk_bf16 = (unsigned*)d_ws;                         // 128 KB
    float*    hcsq     = (float*)((char*)d_ws + CBK_DW * 4);      //   4 KB

    float* out_codes = (float*)d_out;                             // (B,N,D)
    float* out_ids   = out_codes + (size_t)M_TOTAL * DIM;         // (B,N)

    vq_prep_codebook<<<(NUM_NT + 255) / 256, 256, 0, stream>>>(
        codebook, cbk_bf16, hcsq);

    vq_argmin_gather<<<1024, WG_THREADS, LDS_BYTES, stream>>>(
        codes, codebook, cbk_bf16, hcsq, out_codes, out_ids);
}
